// Net_49160195670457
// MI455X (gfx1250) — matrix operand order repaired and checked by reading, not compiled, not run on hardware
//
#include <hip/hip_runtime.h>

// ============================================================================
// MI455X (gfx1250) implementation.
// Trunk convs -> block-cooperative implicit GEMM on v_wmma_f32_16x16x32_bf16:
// each 256-thread block computes a 64(Cout) x 128(B*Lout) tile. Per 32-wide K
// chunk the im2col B tile (32x128 bf16) is gathered once into LDS and shared
// by all 8 waves (4 M-subtiles x 2 N-subtiles); B fragments are 16 strided 2B
// ds loads. Weights are K-padded bf16 so A fragments are two contiguous 16B
// global loads. BN stats/apply, pooling, GATv2 edge softmax, TopK, GAP and
// the MLP head run as fp32 VALU kernels. All scratch in d_ws, deterministic.
// ============================================================================

#define NB 64            // batch
#define MNODES 256       // graph nodes (= trunk channels)
#define LDSW 136         // LDS B-tile row stride in ushorts (272B: 16B-aligned, staggered)

typedef __attribute__((ext_vector_type(16))) unsigned short ushort16;
typedef __attribute__((ext_vector_type(8)))  unsigned short ushort8;
typedef __attribute__((ext_vector_type(16))) __bf16 v16bf;
typedef __attribute__((ext_vector_type(8)))  float  v8f;

__device__ __forceinline__ unsigned short f2bf(float f) {
  unsigned u = __float_as_uint(f);
  unsigned r = (u + 0x7FFFu + ((u >> 16) & 1u)) >> 16;   // round-to-nearest-even
  return (unsigned short)r;
}

// ---------------------------------------------------------------------------
// elementwise / utility
// ---------------------------------------------------------------------------
__global__ void k_cast_bf16(const float* __restrict__ s, unsigned short* __restrict__ d, long n) {
  long i = (long)blockIdx.x * blockDim.x + threadIdx.x;
  if (i < n) d[i] = f2bf(s[i]);
}
__global__ void k_fill_f32(float* p, float v, long n) {
  long i = (long)blockIdx.x * blockDim.x + threadIdx.x;
  if (i < n) p[i] = v;
}
__global__ void k_fill_u32(unsigned* p, unsigned v, long n) {
  long i = (long)blockIdx.x * blockDim.x + threadIdx.x;
  if (i < n) p[i] = v;
}
__global__ void k_bias_bcast(float* __restrict__ out, const float* __restrict__ bias, int total) {
  int i = blockIdx.x * blockDim.x + threadIdx.x;
  if (i < total) out[i] = bias[i & 63];
}
// weight staging: fp32 [Cout, Ktot] -> bf16 [Cout, Kp] with zero K-padding
__global__ void k_stage_w(const float* __restrict__ src, unsigned short* __restrict__ dst,
                          int Ktot, int Kp, int total) {
  int i = blockIdx.x * blockDim.x + threadIdx.x;
  if (i >= total) return;
  int m = i / Kp;
  int k = i - m * Kp;
  dst[i] = (k < Ktot) ? f2bf(src[m * Ktot + k]) : (unsigned short)0;
}

// ---------------------------------------------------------------------------
// Block-cooperative implicit-GEMM conv1d on WMMA bf16.
//   out[b,co,t] = bias[co] + sum_{ci,kw} w[co, ci*KW+kw] * in[b,ci, t*stride+kw-pad]
// Block tile: M=64 (Cout), N=128 (flattened B*Lout), K chunk = 32.
// Waves: waveid&3 -> M subtile (16 rows), waveid>>2 -> N subtile (64 cols,
// i.e. 4 WMMA accumulators per wave).
// Fragment layouts per CDNA5 ISA 7.12.2:
//   A (16x32 bf16): lane&15 = M, kb=8*(lane>>4); e<8: K=kb+e ; e>=8: K=16+kb+e-8
//     -> two contiguous 16B loads from K-padded weights
//   B (32x16 bf16): lane&15 = N; K order as A -> 16 strided 2B LDS reads
//   C/D (16x16 f32): lane&15 = N ; M = 8*(lane>>4) + elem
// NB*Lout is a multiple of 128 (Lout even), and Lout >= 282 > 127, so a block
// N-span is always fully in range and crosses at most one batch boundary.
// ---------------------------------------------------------------------------
__global__ void __launch_bounds__(256) k_conv_wmma(
    const unsigned short* __restrict__ in,
    const unsigned short* __restrict__ w,   // [Cout, Kp] bf16, K-padded
    const float* __restrict__ bias,
    float* __restrict__ out,
    int Cin, int Lin, int Cout, int Lout,
    int KW, int stride, int pad, int Ktot, int Kp) {
  __shared__ __align__(16) unsigned short Bs[32 * LDSW];

  const int tid = threadIdx.x;
  const int lane = tid & 31;
  const int waveid = tid >> 5;
  const int mw = waveid & 3;               // M subtile within block
  const int nw = waveid >> 2;              // N subtile within block (0/1)

  const int Ntot = NB * Lout;
  const int nblk = Ntot >> 7;              // 128-wide N tiles
  const int bm = blockIdx.x / nblk;
  const int bn = blockIdx.x - bm * nblk;
  const int m0 = (bm << 6) + (mw << 4);
  const int n0blk = bn << 7;

  const int b0blk = n0blk / Lout;          // only division in the block
  const int t0blk = n0blk - b0blk * Lout;
  const int base0 = b0blk * Cin * Lin;
  const int base1 = base0 + Cin * Lin;

  const int half = lane >> 4;
  const int kb = half << 3;
  const int lid = lane & 15;

  v8f acc[4];
#pragma unroll
  for (int s = 0; s < 4; ++s) acc[s] = (v8f){0.f,0.f,0.f,0.f,0.f,0.f,0.f,0.f};

  const unsigned short* wrow = w + (m0 + lid) * Kp;
  // cooperative stage mapping: thread covers row krow, columns ngrp..ngrp+15
  const int krow = tid >> 3;               // 0..31
  const int ngrp = (tid & 7) << 4;         // 0,16,...,112

  for (int kc = 0; kc < Ktot; kc += 32) {
    __syncthreads();
    {
      int kg = kc + krow;
      int valid = kg < Ktot;               // only false in conv1 tail chunk
      int ci = valid ? (kg / KW) : 0;
      int kw = kg - ci * KW;
      int cb = ci * Lin;
      unsigned short* dstrow = &Bs[krow * LDSW + ngrp];
#pragma unroll
      for (int e = 0; e < 16; ++e) {
        int toff = t0blk + ngrp + e;
        int wr = toff >= Lout;
        int t = toff - (wr ? Lout : 0);
        int q = t * stride + kw - pad;
        unsigned short v = 0;
        if (valid && q >= 0 && q < Lin)
          v = in[(wr ? base1 : base0) + cb + q];
        dstrow[e] = v;
      }
    }
    __syncthreads();

    // ---- A fragment: two aligned 16B global loads (Kp, kc multiples of 32)
    __builtin_prefetch(wrow + kc + 32, 0, 1);      // global_prefetch_b8
    ushort8 alo = *(const ushort8*)(wrow + kc + kb);
    ushort8 ahi = *(const ushort8*)(wrow + kc + 16 + kb);
    ushort16 av;
#pragma unroll
    for (int e = 0; e < 8; ++e) { av[e] = alo[e]; av[8 + e] = ahi[e]; }
    const v16bf a = __builtin_bit_cast(v16bf, av);

    // ---- B fragments from LDS: lane&15 = N column; e<8: K=kb+e ; e>=8: K=16+kb+e-8 ----
    const unsigned short* brow = &Bs[(nw << 6) + lid];
#pragma unroll
    for (int s = 0; s < 4; ++s) {
      ushort16 bv;
#pragma unroll
      for (int e = 0; e < 8; ++e) { bv[e] = brow[(kb + e) * LDSW + s * 16]; bv[8 + e] = brow[(16 + kb + e) * LDSW + s * 16]; }
      acc[s] = __builtin_amdgcn_wmma_f32_16x16x32_bf16(
          false, a, false, __builtin_bit_cast(v16bf, bv),
          (short)0, acc[s], false, false);
    }
  }

  // ---- store D: lane&15 = N column, rows m0 + kb + e ----
#pragma unroll
  for (int s = 0; s < 4; ++s) {
    int toff = t0blk + (nw << 6) + s * 16 + lid;   // < 2*Lout (Lout >= 282)
    int wr = toff >= Lout;
    int t = toff - (wr ? Lout : 0);
    int b = b0blk + wr;
#pragma unroll
    for (int e = 0; e < 8; ++e) {
      int m = m0 + kb + e;
      out[(b * Cout + m) * Lout + t] = acc[s][e] + bias[m];
    }
  }
}

// ---------------------------------------------------------------------------
// BatchNorm (training-mode batch stats over (N,L)), pooling, residual glue
// ---------------------------------------------------------------------------
__global__ void k_bn_stats(const float* __restrict__ x, float* __restrict__ mv, int C, int L) {
  int c = blockIdx.x, tid = threadIdx.x;
  __shared__ float s1[256], s2[256];
  float a = 0.f, b = 0.f;
  for (int bb = 0; bb < NB; ++bb) {
    const float* p = x + ((long)bb * C + c) * L;
    for (int l = tid; l < L; l += 256) { float v = p[l]; a += v; b += v * v; }
  }
  s1[tid] = a; s2[tid] = b;
  __syncthreads();
  for (int s = 128; s > 0; s >>= 1) {
    if (tid < s) { s1[tid] += s1[tid + s]; s2[tid] += s2[tid + s]; }
    __syncthreads();
  }
  if (tid == 0) {
    float inv = 1.f / ((float)NB * (float)L);
    float m = s1[0] * inv;
    mv[c] = m;
    mv[C + c] = s2[0] * inv - m * m;
  }
}

__global__ void k_bn_apply(float* __restrict__ x, const float* __restrict__ mv,
                           const float* __restrict__ g, const float* __restrict__ bt,
                           int C, int L, int relu, unsigned short* __restrict__ bf) {
  int l = blockIdx.x * blockDim.x + threadIdx.x;
  if (l >= L) return;
  int c = blockIdx.y, b = blockIdx.z;
  long idx = ((long)b * C + c) * L + l;
  float y = (x[idx] - mv[c]) * rsqrtf(mv[C + c] + 1e-5f) * g[c] + bt[c];
  if (relu) y = fmaxf(y, 0.f);
  x[idx] = y;
  if (bf) bf[idx] = f2bf(y);
}

__global__ void k_bn_add_relu(const float* __restrict__ t2, const float* __restrict__ sc,
                              const float* __restrict__ mv, const float* __restrict__ g,
                              const float* __restrict__ bt, float* __restrict__ out,
                              unsigned short* __restrict__ obf, int C, int L) {
  int l = blockIdx.x * blockDim.x + threadIdx.x;
  if (l >= L) return;
  int c = blockIdx.y, b = blockIdx.z;
  long idx = ((long)b * C + c) * L + l;
  float y = (t2[idx] - mv[c]) * rsqrtf(mv[C + c] + 1e-5f) * g[c] + bt[c] + sc[idx];
  y = fmaxf(y, 0.f);
  out[idx] = y;
  obf[idx] = f2bf(y);
}

__global__ void k_maxpool_relu(const float* __restrict__ x, float* __restrict__ out,
                               unsigned short* __restrict__ obf, int Lin, int Lout,
                               int stride, int C) {
  int l = blockIdx.x * blockDim.x + threadIdx.x;
  if (l >= Lout) return;
  int c = blockIdx.y, b = blockIdx.z;
  const float* p = x + ((long)b * C + c) * Lin;
  int p0 = l * stride - 1;
  float m = -3.4e38f;
  for (int j = 0; j < 3; ++j) { int q = p0 + j; if (q >= 0 && q < Lin) m = fmaxf(m, p[q]); }
  m = fmaxf(m, 0.f);
  long o = ((long)b * C + c) * Lout + l;
  out[o] = m; obf[o] = f2bf(m);
}

// sc = 0.5*(maxpool3(x) + avgpool3(x)), pad=1, avg divides by 3 always
__global__ void k_pool_mix(const float* __restrict__ x, float* __restrict__ out,
                           int Lin, int Lout, int stride, int C) {
  int l = blockIdx.x * blockDim.x + threadIdx.x;
  if (l >= Lout) return;
  int c = blockIdx.y, b = blockIdx.z;
  const float* p = x + ((long)b * C + c) * Lin;
  int p0 = l * stride - 1;
  float m = -3.4e38f, su = 0.f;
  for (int j = 0; j < 3; ++j) {
    int q = p0 + j;
    if (q >= 0 && q < Lin) { float v = p[q]; m = fmaxf(m, v); su += v; }
  }
  out[((long)b * C + c) * Lout + l] = 0.5f * (m + su * (1.f / 3.f));
}

__global__ void k_rowmean(const float* __restrict__ h, float* __restrict__ x1, int L, int total) {
  int idx = blockIdx.x * blockDim.x + threadIdx.x;
  if (idx >= total) return;
  const float* p = h + (long)idx * L;
  float s = 0.f;
  for (int l = 0; l < L; ++l) s += p[l];
  x1[idx] = s / (float)L;
}

// ---------------------------------------------------------------------------
// GATv2 + TopK + GAP (per-batch graphs, 256 nodes, fixed edge set + self loops)
// ---------------------------------------------------------------------------
__global__ void k_build_edges(const int* __restrict__ ei, int* __restrict__ src,
                              int* __restrict__ dst, int E0) {
  int i = blockIdx.x * blockDim.x + threadIdx.x;
  if (i < E0) { src[i] = ei[i]; dst[i] = ei[E0 + i]; }
  else if (i < E0 + MNODES) { src[i] = i - E0; dst[i] = i - E0; }
}

__global__ void k_gat_proj(const float* __restrict__ x, const float* __restrict__ Wl,
                           const float* __restrict__ bl, const float* __restrict__ Wr,
                           const float* __restrict__ br, float* __restrict__ xl,
                           float* __restrict__ xr, int Fin, int total) {
  int idx = blockIdx.x * blockDim.x + threadIdx.x;
  if (idx >= total) return;
  int f = idx & 63;
  long node = (long)(idx >> 6);               // b*256+n
  const float* xv = x + node * Fin;
  float al = bl[f], ar = br[f];
  for (int i = 0; i < Fin; ++i) {
    float v = xv[i];
    al += v * Wl[i * 64 + f];
    ar += v * Wr[i * 64 + f];
  }
  xl[idx] = al; xr[idx] = ar;
}

__global__ void k_gat_edge1(const float* __restrict__ xl, const float* __restrict__ xr,
                            const float* __restrict__ att, const int* __restrict__ src,
                            const int* __restrict__ dst, const int* __restrict__ mask,
                            float* __restrict__ esc, unsigned* __restrict__ dmax,
                            int E, int total) {
  int idx = blockIdx.x * blockDim.x + threadIdx.x;
  if (idx >= total) return;
  int b = idx / E, e = idx - b * E;
  int s = src[e], d = dst[e];
  int em = mask[b * MNODES + s] & mask[b * MNODES + d];
  const float* pl = xl + ((long)b * MNODES + s) * 64;
  const float* pr = xr + ((long)b * MNODES + d) * 64;
  float acc = 0.f;
  for (int f = 0; f < 64; ++f) {
    float v = pl[f] + pr[f];
    v = v > 0.f ? v : 0.2f * v;              // leaky_relu(0.2)
    acc += v * att[f];
  }
  if (!em) acc = -1e9f;
  esc[idx] = acc;
  unsigned bits = __float_as_uint(acc);      // order-preserving uint key
  unsigned key = (bits & 0x80000000u) ? ~bits : (bits | 0x80000000u);
  atomicMax(&dmax[b * MNODES + d], key);
}

__global__ void k_gat_edge2(float* __restrict__ esc, const unsigned* __restrict__ dmax,
                            float* __restrict__ den, const int* __restrict__ src,
                            const int* __restrict__ dst, const int* __restrict__ mask,
                            int E, int total) {
  int idx = blockIdx.x * blockDim.x + threadIdx.x;
  if (idx >= total) return;
  int b = idx / E, e = idx - b * E;
  int s = src[e], d = dst[e];
  int em = mask[b * MNODES + s] & mask[b * MNODES + d];
  unsigned key = dmax[b * MNODES + d];
  unsigned bits = (key & 0x80000000u) ? (key & 0x7FFFFFFFu) : ~key;
  float mx = __uint_as_float(bits);
  float ex = em ? expf(esc[idx] - mx) : 0.f;
  esc[idx] = ex;
  atomicAdd(&den[b * MNODES + d], ex);
}

__global__ void k_gat_edge3(const float* __restrict__ esc, const float* __restrict__ den,
                            const float* __restrict__ xl, const int* __restrict__ src,
                            const int* __restrict__ dst, float* __restrict__ out, int E) {
  int be = blockIdx.x;
  int b = be / E, e = be - b * E;
  int f = threadIdx.x;
  int s = src[e], d = dst[e];
  float alpha = esc[be] / (den[b * MNODES + d] + 1e-16f);
  atomicAdd(&out[((long)b * MNODES + d) * 64 + f],
            xl[((long)b * MNODES + s) * 64 + f] * alpha);
}

// score = tanh(h.w/||w||); keep top-k valid nodes (argsort-desc, idx-desc ties);
// h scaled by score for ALL nodes (as in reference).
__global__ void k_topk(float* __restrict__ h, int* __restrict__ mask,
                       const float* __restrict__ w, int kcount) {
  int b = blockIdx.x, n = threadIdx.x;
  __shared__ float sc[MNODES];
  __shared__ int vld[MNODES];
  float nw = 0.f;
  for (int f = 0; f < 64; ++f) nw += w[f] * w[f];
  nw = sqrtf(nw);
  float* hv = h + ((long)b * MNODES + n) * 64;
  float s = 0.f;
  for (int f = 0; f < 64; ++f) s += hv[f] * w[f];
  s = tanhf(s / nw);
  sc[n] = s; vld[n] = mask[b * MNODES + n];
  __syncthreads();
  int rank = 0;
  for (int j = 0; j < MNODES; ++j) {
    if (vld[j]) {
      float sj = sc[j];
      if (sj > s || (sj == s && j > n)) rank++;
    }
  }
  for (int f = 0; f < 64; ++f) hv[f] *= s;
  mask[b * MNODES + n] = (vld[n] && rank < kcount) ? 1 : 0;
}

__global__ void k_gap(const float* __restrict__ h, const int* __restrict__ mask,
                      float* __restrict__ x2) {
  int b = blockIdx.x, f = threadIdx.x;
  float s = 0.f; int cnt = 0;
  for (int n = 0; n < MNODES; ++n)
    if (mask[b * MNODES + n]) { s += h[((long)b * MNODES + n) * 64 + f]; cnt++; }
  x2[b * 64 + f] += s / (float)cnt;
}

// ---------------------------------------------------------------------------
// MLP head: 320 -> 128 -> 32 -> 8 -> 3 (relu between), per batch row
// ---------------------------------------------------------------------------
__global__ void k_head(const float* __restrict__ x1, const float* __restrict__ x2,
                       const float* __restrict__ w1, const float* __restrict__ b1,
                       const float* __restrict__ w2, const float* __restrict__ b2,
                       const float* __restrict__ w3, const float* __restrict__ b3,
                       const float* __restrict__ w4, const float* __restrict__ b4,
                       float* __restrict__ out) {
  int b = blockIdx.x, t = threadIdx.x;
  __shared__ float z[320], z1[128], z2[32], z3[8];
  for (int i = t; i < 256; i += 128) z[i] = x1[b * 256 + i];
  if (t < 64) z[256 + t] = x2[b * 64 + t];
  __syncthreads();
  if (t < 128) { float a = b1[t]; for (int i = 0; i < 320; ++i) a += z[i] * w1[i * 128 + t]; z1[t] = fmaxf(a, 0.f); }
  __syncthreads();
  if (t < 32)  { float a = b2[t]; for (int i = 0; i < 128; ++i) a += z1[i] * w2[i * 32 + t]; z2[t] = fmaxf(a, 0.f); }
  __syncthreads();
  if (t < 8)   { float a = b3[t]; for (int i = 0; i < 32; ++i)  a += z2[i] * w3[i * 8 + t];  z3[t] = fmaxf(a, 0.f); }
  __syncthreads();
  if (t < 3)   { float a = b4[t]; for (int i = 0; i < 8; ++i)   a += z3[i] * w4[i * 3 + t];  out[b * 3 + t] = a; }
}

// ============================================================================
// Host orchestration
// ============================================================================
// Input flattening assumed = jax pytree order (dict keys sorted at every level):
//  0 edge_index | 1..24 gnn.c11/c22/c33/c44 {Wl,Wr,att,bias,bl,br} | 25..28 w11..w44
//  29..36 head {b1..b4,w1..w4} | 37..40 trunk {b1,c1b,c1w,g1}
//  41.. r21,r22,r31,r32,r41,r42,r51,r52 each {b1,b2,c1b,c1w,c2b,c2w,c3b,c3w,g1,g2}
//  121 x
enum {
  IDX_EDGE = 0,
  IDX_C11 = 1, IDX_C22 = 7, IDX_C33 = 13, IDX_C44 = 19,
  IDX_W11 = 25, IDX_W22 = 26, IDX_W33 = 27, IDX_W44 = 28,
  IDX_HB1 = 29, IDX_HB2 = 30, IDX_HB3 = 31, IDX_HB4 = 32,
  IDX_HW1 = 33, IDX_HW2 = 34, IDX_HW3 = 35, IDX_HW4 = 36,
  IDX_TB1 = 37, IDX_TC1B = 38, IDX_TC1W = 39, IDX_TG1 = 40,
  IDX_R21 = 41, IDX_R22 = 51, IDX_R31 = 61, IDX_R32 = 71,
  IDX_R41 = 81, IDX_R42 = 91, IDX_R51 = 101, IDX_R52 = 111,
  IDX_X = 121
};
// residual-block member offsets (sorted keys)
enum { RB_B1 = 0, RB_B2 = 1, RB_C1B = 2, RB_C1W = 3, RB_C2B = 4, RB_C2W = 5,
       RB_C3B = 6, RB_C3W = 7, RB_G1 = 8, RB_G2 = 9 };
// gat-params member offsets (sorted keys)
enum { GP_WL = 0, GP_WR = 1, GP_ATT = 2, GP_BIAS = 3, GP_BL = 4, GP_BR = 5 };

extern "C" void kernel_launch(void* const* d_in, const int* in_sizes, int n_in,
                              void* d_out, int out_size, void* d_ws, size_t ws_size,
                              hipStream_t stream) {
  (void)n_in; (void)out_size; (void)ws_size;

  // ---- workspace bump allocator (deterministic) ----
  char* base = (char*)d_ws;
  size_t off = 0;
  auto alloc = [&](size_t bytes) -> char* {
    off = (off + 255) & ~(size_t)255;
    char* p = base + off;
    off += bytes;
    return p;
  };
  const int E0 = in_sizes[IDX_EDGE] / 2;
  const int E  = E0 + MNODES;

  float* F0 = (float*)alloc((size_t)18481152 * 4);       // conv1 out / t2
  float* F1 = (float*)alloc((size_t)9240576 * 4);        // current activation
  float* F2 = (float*)alloc((size_t)9240576 * 4);        // t1
  float* F3 = (float*)alloc((size_t)9240576 * 4);        // shortcut
  unsigned short* H0 = (unsigned short*)alloc((size_t)9240576 * 2);
  unsigned short* H1 = (unsigned short*)alloc((size_t)9240576 * 2);
  unsigned short* H2 = (unsigned short*)alloc((size_t)9240576 * 2);
  unsigned short* WBF = (unsigned short*)alloc((size_t)2100000 * 2);
  float* MV  = (float*)alloc(512 * 4);
  float* MV2 = (float*)alloc(512 * 4);
  float* X1  = (float*)alloc((size_t)NB * 256 * 4);
  float* X2  = (float*)alloc((size_t)NB * 64 * 4);
  float* HGA = (float*)alloc((size_t)NB * 256 * 64 * 4);
  float* HGB = (float*)alloc((size_t)NB * 256 * 64 * 4);
  float* XL  = (float*)alloc((size_t)NB * 256 * 64 * 4);
  float* XR  = (float*)alloc((size_t)NB * 256 * 64 * 4);
  float* ESC = (float*)alloc((size_t)NB * E * 4);
  unsigned* DMAX = (unsigned*)alloc((size_t)NB * 256 * 4);
  float* DEN = (float*)alloc((size_t)NB * 256 * 4);
  int* MASK = (int*)alloc((size_t)NB * 256 * 4);
  int* SRC  = (int*)alloc((size_t)E * 4);
  int* DST  = (int*)alloc((size_t)E * 4);

  auto IN = [&](int i) { return (const float*)d_in[i]; };
  auto cast = [&](const float* s, unsigned short* d, long n) {
    k_cast_bf16<<<dim3((unsigned)((n + 255) / 256)), 256, 0, stream>>>(s, d, n);
  };
  // stage conv weight [Cout, Ktot] -> bf16 [Cout, Kp], Kp = roundup(Ktot, 32)
  size_t woff = 0;
  auto stagew = [&](int idx, int Cout, int Ktot, int* KpOut) -> const unsigned short* {
    int Kp = (Ktot + 31) & ~31;
    unsigned short* d = WBF + woff;
    int total = Cout * Kp;
    k_stage_w<<<(total + 255) / 256, 256, 0, stream>>>(IN(idx), d, Ktot, Kp, total);
    woff += (size_t)total;
    *KpOut = Kp;
    return d;
  };
  auto conv = [&](const unsigned short* in, const unsigned short* wbf, const float* bias,
                  float* out, int Cin, int Lin, int Cout, int Lout, int KW, int stride,
                  int pad, int Kp) {
    unsigned blocks = (unsigned)(((long)NB * Lout >> 7) * (Cout >> 6));  // exact tiling
    k_conv_wmma<<<blocks, 256, 0, stream>>>(in, wbf, bias, out, Cin, Lin, Cout, Lout,
                                            KW, stride, pad, Cin * KW, Kp);
  };
  auto stats = [&](const float* x, float* mv, int C, int L) {
    k_bn_stats<<<C, 256, 0, stream>>>(x, mv, C, L);
  };

  // ---------------- trunk ----------------
  cast(IN(IDX_X), H0, (long)NB * 7 * 9024);
  int kp;
  const unsigned short* wc1 = stagew(IDX_TC1W, 64, 49, &kp);
  conv(H0, wc1, IN(IDX_TC1B), F0, 7, 9024, 64, 4512, 7, 2, 3, kp);
  stats(F0, MV, 64, 4512);
  { dim3 g((4512 + 255) / 256, 64, NB);
    k_bn_apply<<<g, 256, 0, stream>>>(F0, MV, IN(IDX_TG1), IN(IDX_TB1), 64, 4512, 1, nullptr); }
  { dim3 g((2256 + 255) / 256, 64, NB);
    k_maxpool_relu<<<g, 256, 0, stream>>>(F0, F1, H0, 4512, 2256, 2, 64); }

  auto residual = [&](int rb, int Cin, int Cout, int Lin, int stride) {
    int Lout = (Lin - 1) / stride + 1;                   // k=3, pad=1
    int kp1, kp2, kp3;
    const unsigned short* w1 = stagew(rb + RB_C1W, Cout, Cin * 3, &kp1);
    const unsigned short* w2 = stagew(rb + RB_C2W, Cout, Cout * 3, &kp2);
    const unsigned short* w3 = stagew(rb + RB_C3W, Cout, Cin, &kp3);
    conv(H0, w1, IN(rb + RB_C1B), F2, Cin, Lin, Cout, Lout, 3, stride, 1, kp1);
    stats(F2, MV, Cout, Lout);
    { dim3 g((Lout + 255) / 256, Cout, NB);
      k_bn_apply<<<g, 256, 0, stream>>>(F2, MV, IN(rb + RB_G1), IN(rb + RB_B1), Cout, Lout, 1, H1); }
    conv(H1, w2, IN(rb + RB_C2B), F0, Cout, Lout, Cout, Lout, 3, 1, 1, kp2);
    stats(F0, MV2, Cout, Lout);
    { dim3 g((Lout + 255) / 256, Cin, NB);
      k_pool_mix<<<g, 256, 0, stream>>>(F1, F3, Lin, Lout, stride, Cin); }
    cast(F3, H2, (long)NB * Cin * Lout);
    conv(H2, w3, IN(rb + RB_C3B), F3, Cin, Lout, Cout, Lout, 1, 1, 0, kp3);
    { dim3 g((Lout + 255) / 256, Cout, NB);
      k_bn_add_relu<<<g, 256, 0, stream>>>(F0, F3, MV2, IN(rb + RB_G2), IN(rb + RB_B2), F1, H0, Cout, Lout); }
  };

  residual(IDX_R21,  64,  64, 2256, 1);
  residual(IDX_R22,  64,  64, 2256, 1);
  residual(IDX_R31,  64, 128, 2256, 2);
  residual(IDX_R32, 128, 128, 1128, 1);
  residual(IDX_R41, 128, 256, 1128, 2);
  residual(IDX_R42, 256, 256,  564, 1);
  residual(IDX_R51, 256, 256,  564, 2);
  residual(IDX_R52, 256, 256,  282, 1);
  // F1 = trunk output [NB, 256, 282]

  k_rowmean<<<(NB * 256 + 255) / 256, 256, 0, stream>>>(F1, X1, 282, NB * 256);

  // ---------------- GNN ----------------
  k_fill_u32<<<(NB * 256 + 255) / 256, 256, 0, stream>>>((unsigned*)MASK, 1u, (long)NB * 256);
  k_fill_f32<<<(NB * 64 + 255) / 256, 256, 0, stream>>>(X2, 0.f, (long)NB * 64);
  k_build_edges<<<(E + 255) / 256, 256, 0, stream>>>((const int*)d_in[IDX_EDGE], SRC, DST, E0);

  const int tot = NB * 256 * 64;
  const int te = NB * E;
  auto gat = [&](const float* xin, int Fin, int cb, float* hout) {
    k_gat_proj<<<tot / 256, 256, 0, stream>>>(xin, IN(cb + GP_WL), IN(cb + GP_BL),
                                              IN(cb + GP_WR), IN(cb + GP_BR), XL, XR, Fin, tot);
    k_fill_u32<<<(NB * 256 + 255) / 256, 256, 0, stream>>>(DMAX, 0u, (long)NB * 256);
    k_fill_f32<<<(NB * 256 + 255) / 256, 256, 0, stream>>>(DEN, 0.f, (long)NB * 256);
    k_bias_bcast<<<tot / 256, 256, 0, stream>>>(hout, IN(cb + GP_BIAS), tot);
    k_gat_edge1<<<(te + 255) / 256, 256, 0, stream>>>(XL, XR, IN(cb + GP_ATT), SRC, DST, MASK, ESC, DMAX, E, te);
    k_gat_edge2<<<(te + 255) / 256, 256, 0, stream>>>(ESC, DMAX, DEN, SRC, DST, MASK, E, te);
    k_gat_edge3<<<te, 64, 0, stream>>>(ESC, DEN, XL, SRC, DST, hout, E);
  };

  gat(F1, 282, IDX_C11, HGA);
  k_topk<<<NB, 256, 0, stream>>>(HGA, MASK, IN(IDX_W11), 205);
  gat(HGA, 64, IDX_C22, HGB);
  k_topk<<<NB, 256, 0, stream>>>(HGB, MASK, IN(IDX_W22), 164);
  k_gap<<<NB, 64, 0, stream>>>(HGB, MASK, X2);
  gat(HGB, 64, IDX_C33, HGA);
  k_topk<<<NB, 256, 0, stream>>>(HGA, MASK, IN(IDX_W33), 132);
  k_gap<<<NB, 64, 0, stream>>>(HGA, MASK, X2);
  gat(HGA, 64, IDX_C44, HGB);
  k_topk<<<NB, 256, 0, stream>>>(HGB, MASK, IN(IDX_W44), 106);
  k_gap<<<NB, 64, 0, stream>>>(HGB, MASK, X2);

  // ---------------- head ----------------
  k_head<<<NB, 128, 0, stream>>>(X1, X2,
                                 IN(IDX_HW1), IN(IDX_HB1), IN(IDX_HW2), IN(IDX_HB2),
                                 IN(IDX_HW3), IN(IDX_HB3), IN(IDX_HW4), IN(IDX_HB4),
                                 (float*)d_out);
}
